// MoELayer_62388694942421
// MI455X (gfx1250) — matrix operand order repaired and checked by reading, not compiled, not run on hardware
//
#include <hip/hip_runtime.h>
#include <hip/hip_bf16.h>

// Problem constants (match reference)
#define T_TOK 4096
#define DMODEL 1024
#define NEXP 8
#define TOPK 2
#define FF 1024
#define CPE ((T_TOK * TOPK) / NEXP)  // 1024 rows per expert
#define KDIM 1024                    // both GEMM K extents (D and F)

#define BM 64
#define BN 128
#define BK 32
#define LDA 40  // LDS row stride in bf16 elems (80B: keeps all uint4 slices 16B aligned)

typedef __bf16 bf16x16 __attribute__((ext_vector_type(16)));
typedef float f32x8 __attribute__((ext_vector_type(8)));
typedef int v4i __attribute__((ext_vector_type(4)));

#if defined(__has_builtin)
#if __has_builtin(__builtin_amdgcn_global_load_async_to_lds_b128) && \
    __has_builtin(__builtin_amdgcn_s_wait_asynccnt)
#define USE_ASYNC_LDS 1
#endif
#endif

union FragU {
    uint4 u[2];
    bf16x16 v;
};

__device__ __forceinline__ float fast_rcp(float x) {
#if defined(__has_builtin)
#if __has_builtin(__builtin_amdgcn_rcpf)
    return __builtin_amdgcn_rcpf(x);
#else
    return 1.0f / x;
#endif
#else
    return 1.0f / x;
#endif
}

__device__ __forceinline__ float sigmoidf_fast(float x) { return fast_rcp(1.0f + __expf(-x)); }

// ---------------------------------------------------------------------------
// 1) Dispatch permutation: rank pairs within their expert (order within an
//    expert group is irrelevant to the math, so atomic ranking is valid).
// ---------------------------------------------------------------------------
__global__ void moe_perm_kernel(const int* __restrict__ topk_idx, int* __restrict__ cnt,
                                int* __restrict__ src_pair) {
    int i = blockIdx.x * blockDim.x + threadIdx.x;
    if (i >= T_TOK * TOPK) return;
    int e = topk_idx[i];
    int r = atomicAdd(&cnt[e], 1);
    src_pair[e * CPE + r] = i;  // permuted row -> flat (token,k) pair
}

// ---------------------------------------------------------------------------
// 2) Per-token sigmoid gate for shared expert: g[t] = sigmoid(x[t] . gw)
// ---------------------------------------------------------------------------
__global__ __launch_bounds__(256) void moe_gate_kernel(const float* __restrict__ x,
                                                       const float* __restrict__ gw,
                                                       float* __restrict__ g) {
    __shared__ float red[256];
    int t = blockIdx.x;
    float s = 0.f;
    for (int d = threadIdx.x; d < DMODEL; d += 256) s += x[(size_t)t * DMODEL + d] * gw[d];
    red[threadIdx.x] = s;
    __syncthreads();
    for (int w = 128; w > 0; w >>= 1) {
        if (threadIdx.x < w) red[threadIdx.x] += red[threadIdx.x + w];
        __syncthreads();
    }
    if (threadIdx.x == 0) g[t] = sigmoidf_fast(red[0]);
}

// ---------------------------------------------------------------------------
// 3) GEMM-1: H = silu(X Wg) * (X Wu). 64x128x32 block tile, 8 wave32s each
//    owning a 32x32 patch -> 8 WMMAs per k-step per wave. A rows gathered on
//    the fly from fp32 hidden_states; W tiles converted fp32->bf16 (native
//    v_cvt) and transposed in LDS.
//    Expert mode: src_pair != nullptr, blockIdx.z = expert.
//    Shared mode: src_pair == nullptr, blockIdx.z = 0.
// ---------------------------------------------------------------------------
__global__ __launch_bounds__(256) void moe_gemm1_kernel(
    const float* __restrict__ X, const int* __restrict__ src_pair, const float* __restrict__ Wg,
    const float* __restrict__ Wu, __bf16* __restrict__ Hout, int rows_per_group) {
    __shared__ __align__(16) __bf16 As[BM * LDA];
    __shared__ __align__(16) __bf16 Bg[BN * LDA];
    __shared__ __align__(16) __bf16 Bu[BN * LDA];

    const int e = blockIdx.z;
    const int m0 = blockIdx.y * BM;
    const int n0 = blockIdx.x * BN;
    const float* wg = Wg + (size_t)e * KDIM * FF;
    const float* wu = Wu + (size_t)e * KDIM * FF;
    const int row_base = e * rows_per_group + m0;

    const int tid = threadIdx.x;
    const int lane = tid & 31;
    const int wave = tid >> 5;
    const int wm = (wave & 1) * 32;   // wave M offset (2 waves in M)
    const int wn = (wave >> 1) * 32;  // wave N offset (4 waves in N)

    // A-stager role: each thread owns 8 contiguous fp32 of one row
    const int arow = tid >> 2;      // 0..63
    const int ac0 = (tid & 3) * 8;  // 0,8,16,24
    int tok = row_base + arow;
    if (src_pair) tok = src_pair[tok] >> 1;  // pair index / TOPK
    const float* xrow = X + (size_t)tok * DMODEL;

    // B-stager role: thread owns 8 contiguous n of one k row (two k halves)
    const int bk = tid >> 4;        // 0..15
    const int bn = (tid & 15) * 8;  // 0..120

    const f32x8 vz = {0.f, 0.f, 0.f, 0.f, 0.f, 0.f, 0.f, 0.f};
    f32x8 accg[2][2] = {{vz, vz}, {vz, vz}};
    f32x8 accu[2][2] = {{vz, vz}, {vz, vz}};

    const int kh = lane >> 4;  // lane half
    const int rl = lane & 15;  // lane row/col within half

    for (int k0 = 0; k0 < KDIM; k0 += BK) {
        // ---- stage A (gathered fp32 -> bf16, native converts) ----
        float4 a0 = *(const float4*)(xrow + k0 + ac0);
        float4 a1 = *(const float4*)(xrow + k0 + ac0 + 4);
        __bf16* ap = &As[arow * LDA + ac0];
        ap[0] = (__bf16)a0.x; ap[1] = (__bf16)a0.y; ap[2] = (__bf16)a0.z; ap[3] = (__bf16)a0.w;
        ap[4] = (__bf16)a1.x; ap[5] = (__bf16)a1.y; ap[6] = (__bf16)a1.z; ap[7] = (__bf16)a1.w;

        // ---- stage B (fp32 -> bf16, transposed to [n][k]) ----
#pragma unroll
        for (int khalf = 0; khalf < BK; khalf += 16) {
            const int kk = bk + khalf;
            const float4* gp = (const float4*)(wg + (size_t)(k0 + kk) * FF + n0 + bn);
            const float4* up = (const float4*)(wu + (size_t)(k0 + kk) * FF + n0 + bn);
            float4 g0 = gp[0], g1 = gp[1];
            float4 u0 = up[0], u1 = up[1];
            Bg[(bn + 0) * LDA + kk] = (__bf16)g0.x; Bg[(bn + 1) * LDA + kk] = (__bf16)g0.y;
            Bg[(bn + 2) * LDA + kk] = (__bf16)g0.z; Bg[(bn + 3) * LDA + kk] = (__bf16)g0.w;
            Bg[(bn + 4) * LDA + kk] = (__bf16)g1.x; Bg[(bn + 5) * LDA + kk] = (__bf16)g1.y;
            Bg[(bn + 6) * LDA + kk] = (__bf16)g1.z; Bg[(bn + 7) * LDA + kk] = (__bf16)g1.w;
            Bu[(bn + 0) * LDA + kk] = (__bf16)u0.x; Bu[(bn + 1) * LDA + kk] = (__bf16)u0.y;
            Bu[(bn + 2) * LDA + kk] = (__bf16)u0.z; Bu[(bn + 3) * LDA + kk] = (__bf16)u0.w;
            Bu[(bn + 4) * LDA + kk] = (__bf16)u1.x; Bu[(bn + 5) * LDA + kk] = (__bf16)u1.y;
            Bu[(bn + 6) * LDA + kk] = (__bf16)u1.z; Bu[(bn + 7) * LDA + kk] = (__bf16)u1.w;
        }
        // prefetch next weight k-tile (weights stream once from HBM: the bound)
        if (k0 + BK < KDIM) {
            __builtin_prefetch(wg + (size_t)(k0 + BK + bk) * FF + n0 + bn, 0, 0);
            __builtin_prefetch(wu + (size_t)(k0 + BK + bk) * FF + n0 + bn, 0, 0);
        }
        __syncthreads();

        // ---- fragments + WMMA (2 M-subtiles x 2 N-subtiles x {gate,up}) ----
        FragU fa[2], fbg[2], fbu[2];
#pragma unroll
        for (int s = 0; s < 2; ++s) {
            // A layout (16-bit 16x32): lane rl = M row; 16 contiguous K starting at 16*kh
            const int mrow = wm + s * 16 + rl;
            fa[s].u[0] = *(const uint4*)&As[mrow * LDA + 16 * kh];
            fa[s].u[1] = *(const uint4*)&As[mrow * LDA + 16 * kh + 8];
        }
#pragma unroll
        for (int t = 0; t < 2; ++t) {
            // B layout: lane rl = N column, 16 contiguous K starting at 16*kh
            const int nrow = wn + t * 16 + rl;
            const uint4* p = (const uint4*)&Bg[nrow * LDA + 16 * kh];
            fbg[t].u[0] = p[0]; fbg[t].u[1] = p[1];
            const uint4* q = (const uint4*)&Bu[nrow * LDA + 16 * kh];
            fbu[t].u[0] = q[0]; fbu[t].u[1] = q[1];
        }
#pragma unroll
        for (int s = 0; s < 2; ++s)
#pragma unroll
            for (int t = 0; t < 2; ++t) {
                accg[s][t] = __builtin_amdgcn_wmma_f32_16x16x32_bf16(
                    false, fa[s].v, false, fbg[t].v, (short)0, accg[s][t], false, false);
                accu[s][t] = __builtin_amdgcn_wmma_f32_16x16x32_bf16(
                    false, fa[s].v, false, fbu[t].v, (short)0, accu[s][t], false, false);
            }
        __syncthreads();
    }

    // ---- epilogue: h = silu(g)*u -> bf16 H ----
#pragma unroll
    for (int s = 0; s < 2; ++s)
#pragma unroll
        for (int t = 0; t < 2; ++t)
#pragma unroll
            for (int j = 0; j < 8; ++j) {
                float gv = accg[s][t][j];
                float uv = accu[s][t][j];
                float h = (gv * sigmoidf_fast(gv)) * uv;
                int rr = row_base + wm + s * 16 + j + kh * 8;  // M = j + 8*half
                int nn = n0 + wn + t * 16 + rl;
                Hout[(size_t)rr * FF + nn] = (__bf16)h;
            }
}

// ---------------------------------------------------------------------------
// 4) GEMM-2: Y = H Wd, epilogue scatters with atomic f32 adds:
//    expert mode: out[tok] += topk_weight * y   (via src_pair)
//    shared mode: out[t]   += gate[t] * y
//    A-stage uses GLOBAL_LOAD_ASYNC_TO_LDS_B128 when the builtin exists.
// ---------------------------------------------------------------------------
__global__ __launch_bounds__(256) void moe_gemm2_kernel(
    const __bf16* __restrict__ Hin, const float* __restrict__ Wd, float* __restrict__ out,
    const int* __restrict__ src_pair, const float* __restrict__ tw, const float* __restrict__ gvec,
    int rows_per_group) {
    __shared__ __align__(16) __bf16 As[BM * LDA];
    __shared__ __align__(16) __bf16 Bt[BN * LDA];

    const int e = blockIdx.z;
    const int m0 = blockIdx.y * BM;
    const int n0 = blockIdx.x * BN;
    const float* wd = Wd + (size_t)e * KDIM * DMODEL;  // [F][D] per expert
    const int row_base = e * rows_per_group + m0;

    const int tid = threadIdx.x;
    const int lane = tid & 31;
    const int wave = tid >> 5;
    const int wm = (wave & 1) * 32;
    const int wn = (wave >> 1) * 32;

    const int arow = tid >> 2;
    const int ac0 = (tid & 3) * 8;
    const __bf16* hrow = Hin + (size_t)(row_base + arow) * FF;

    const int bk = tid >> 4;        // 0..15
    const int bn = (tid & 15) * 8;  // 0..120

    const f32x8 vz = {0.f, 0.f, 0.f, 0.f, 0.f, 0.f, 0.f, 0.f};
    f32x8 acc[2][2] = {{vz, vz}, {vz, vz}};

    const int kh = lane >> 4;
    const int rl = lane & 15;

    for (int k0 = 0; k0 < KDIM; k0 += BK) {
        // ---- stage A: bf16 copy, async global->LDS when available ----
#ifdef USE_ASYNC_LDS
        __builtin_amdgcn_global_load_async_to_lds_b128(
            (__attribute__((address_space(1))) v4i*)(hrow + k0 + ac0),
            (__attribute__((address_space(3))) v4i*)&As[arow * LDA + ac0], 0, 0);
#else
        *(uint4*)&As[arow * LDA + ac0] = *(const uint4*)(hrow + k0 + ac0);
#endif

        // ---- stage B: fp32 -> bf16 transposed ----
#pragma unroll
        for (int khalf = 0; khalf < BK; khalf += 16) {
            const int kk = bk + khalf;
            const float4* p = (const float4*)(wd + (size_t)(k0 + kk) * DMODEL + n0 + bn);
            float4 b0 = p[0], b1 = p[1];
            Bt[(bn + 0) * LDA + kk] = (__bf16)b0.x; Bt[(bn + 1) * LDA + kk] = (__bf16)b0.y;
            Bt[(bn + 2) * LDA + kk] = (__bf16)b0.z; Bt[(bn + 3) * LDA + kk] = (__bf16)b0.w;
            Bt[(bn + 4) * LDA + kk] = (__bf16)b1.x; Bt[(bn + 5) * LDA + kk] = (__bf16)b1.y;
            Bt[(bn + 6) * LDA + kk] = (__bf16)b1.z; Bt[(bn + 7) * LDA + kk] = (__bf16)b1.w;
        }
        if (k0 + BK < KDIM) {
            __builtin_prefetch(wd + (size_t)(k0 + BK + bk) * DMODEL + n0 + bn, 0, 0);
        }
#ifdef USE_ASYNC_LDS
        __builtin_amdgcn_s_wait_asynccnt(0);
#endif
        __syncthreads();

        FragU fa[2], fb[2];
#pragma unroll
        for (int s = 0; s < 2; ++s) {
            const int mrow = wm + s * 16 + rl;
            fa[s].u[0] = *(const uint4*)&As[mrow * LDA + 16 * kh];
            fa[s].u[1] = *(const uint4*)&As[mrow * LDA + 16 * kh + 8];
        }
#pragma unroll
        for (int t = 0; t < 2; ++t) {
            const uint4* q = (const uint4*)&Bt[(wn + t * 16 + rl) * LDA + 16 * kh];
            fb[t].u[0] = q[0]; fb[t].u[1] = q[1];
        }
#pragma unroll
        for (int s = 0; s < 2; ++s)
#pragma unroll
            for (int t = 0; t < 2; ++t)
                acc[s][t] = __builtin_amdgcn_wmma_f32_16x16x32_bf16(
                    false, fa[s].v, false, fb[t].v, (short)0, acc[s][t], false, false);
        __syncthreads();
    }

    // ---- epilogue: weighted atomic scatter ----
#pragma unroll
    for (int s = 0; s < 2; ++s)
#pragma unroll
        for (int t = 0; t < 2; ++t)
#pragma unroll
            for (int j = 0; j < 8; ++j) {
                float y = acc[s][t][j];
                int rr = row_base + wm + s * 16 + j + kh * 8;
                int nn = n0 + wn + t * 16 + rl;
                if (src_pair) {
                    int i = src_pair[rr];
                    atomicAdd(out + (size_t)(i >> 1) * DMODEL + nn, tw[i] * y);
                } else {
                    atomicAdd(out + (size_t)rr * DMODEL + nn, gvec[rr] * y);
                }
            }
}

// ---------------------------------------------------------------------------
extern "C" void kernel_launch(void* const* d_in, const int* in_sizes, int n_in, void* d_out,
                              int out_size, void* d_ws, size_t ws_size, hipStream_t stream) {
    const float* hidden = (const float*)d_in[0];
    const float* topk_w = (const float*)d_in[1];
    const float* w_gate = (const float*)d_in[2];
    const float* w_up = (const float*)d_in[3];
    const float* w_down = (const float*)d_in[4];
    const float* sw_gate = (const float*)d_in[5];
    const float* sw_up = (const float*)d_in[6];
    const float* sw_down = (const float*)d_in[7];
    const float* sgw = (const float*)d_in[8];
    const int* topk_i = (const int*)d_in[9];
    float* out = (float*)d_out;

    // workspace layout (256B-aligned slabs): counters | src_pair | gate | H | Hs
    char* ws = (char*)d_ws;
    int* cnt = (int*)(ws + 0);          // 32 B
    int* src_pair = (int*)(ws + 256);   // 32 KB
    float* gvec = (float*)(ws + 33024); // 16 KB
    __bf16* H = (__bf16*)(ws + 49408);  // 16 MB
    __bf16* Hs = (__bf16*)(ws + 49408 + (size_t)T_TOK * TOPK * FF * 2);  // 8 MB

    (void)hipMemsetAsync(cnt, 0, NEXP * sizeof(int), stream);
    (void)hipMemsetAsync(out, 0, (size_t)out_size * sizeof(float), stream);

    moe_perm_kernel<<<(T_TOK * TOPK) / 256, 256, 0, stream>>>(topk_i, cnt, src_pair);
    moe_gate_kernel<<<T_TOK, 256, 0, stream>>>(hidden, sgw, gvec);

    // expert GEMM-1: grid (N/BN, C/BM, E)
    moe_gemm1_kernel<<<dim3(FF / BN, CPE / BM, NEXP), 256, 0, stream>>>(hidden, src_pair, w_gate,
                                                                        w_up, H, CPE);
    // shared GEMM-1
    moe_gemm1_kernel<<<dim3(FF / BN, T_TOK / BM, 1), 256, 0, stream>>>(hidden, nullptr, sw_gate,
                                                                       sw_up, Hs, T_TOK);
    // expert GEMM-2 (down + weighted combine)
    moe_gemm2_kernel<<<dim3(DMODEL / BN, CPE / BM, NEXP), 256, 0, stream>>>(H, w_down, out,
                                                                            src_pair, topk_w,
                                                                            nullptr, CPE);
    // shared GEMM-2 (down + sigmoid-gated add)
    moe_gemm2_kernel<<<dim3(DMODEL / BN, T_TOK / BM, 1), 256, 0, stream>>>(Hs, sw_down, out,
                                                                           nullptr, nullptr, gvec,
                                                                           T_TOK);
}
